// HiggsLinear_46188078301680
// MI455X (gfx1250) — matrix operand order repaired and checked by reading, not compiled, not run on hardware
//
#include <hip/hip_runtime.h>
#include <hip/hip_bf16.h>

typedef __attribute__((ext_vector_type(16))) _Float16 v16h;
typedef __attribute__((ext_vector_type(8)))  _Float16 v8h;
typedef __attribute__((ext_vector_type(2)))  _Float16 v2h;
typedef __attribute__((ext_vector_type(8)))  float    v8f;
typedef __attribute__((ext_vector_type(4)))  int      v4i;

#if __has_builtin(__builtin_amdgcn_global_load_async_to_lds_b128) && \
    __has_builtin(__builtin_amdgcn_s_wait_asynccnt)
#define HAS_ASYNC 1
#else
#define HAS_ASYNC 0
#endif

#define HAD  1024
#define IN_F 4096

// ---------------------------------------------------------------------------
// Kernel 1: blockwise 1024-point fast Hadamard transform of x, f32 -> f16.
// ---------------------------------------------------------------------------
__global__ void __launch_bounds__(256) had_kernel(const float* __restrict__ x,
                                                  _Float16* __restrict__ xh) {
  __shared__ float buf[HAD];
  const int t = threadIdx.x;
  const size_t base = (size_t)blockIdx.x * HAD;  // flat over (token, block)
#pragma unroll
  for (int i = 0; i < 4; ++i) buf[t + i * 256] = x[base + t + i * 256];
  __syncthreads();
#pragma unroll
  for (int s = 0; s < 10; ++s) {
#pragma unroll
    for (int j = 0; j < 2; ++j) {
      int p = t + j * 256;                                  // 512 pairs/stage
      int i = ((p >> s) << (s + 1)) | (p & ((1 << s) - 1));
      float a = buf[i];
      float b = buf[i + (1 << s)];
      buf[i] = a + b;
      buf[i + (1 << s)] = a - b;
    }
    __syncthreads();
  }
#pragma unroll
  for (int i = 0; i < 4; ++i) {
    int idx = t + i * 256;
    xh[base + idx] = (_Float16)(buf[idx] * 0.03125f);
  }
}

// ---------------------------------------------------------------------------
// Kernel 2: dequantize weights (codebook gather + per-1024-group scale) -> f16
// ---------------------------------------------------------------------------
__global__ void __launch_bounds__(256) dequant_kernel(const int* __restrict__ codes,
                                                      const float* __restrict__ grid,
                                                      const float* __restrict__ scales,
                                                      _Float16* __restrict__ Wd) {
  size_t idx = (size_t)blockIdx.x * 256 + threadIdx.x;   // o*2048 + g
  int o = (int)(idx >> 11);
  int g = (int)(idx & 2047);
  int code = codes[idx];
  float s = scales[(o << 2) | (g >> 9)];                 // (2g)/1024 == g>>9
  v2h w;
  w.x = (_Float16)(grid[code * 2 + 0] * s);
  w.y = (_Float16)(grid[code * 2 + 1] * s);
  *(v2h*)(Wd + idx * 2) = w;
}

// ---------------------------------------------------------------------------
// Kernel 3: out = xh @ Wd^T + bias, v_wmma_f32_16x16x32_f16.
// Block tile 128x128, BK=32, 8 waves (2Mx4N), wave tile 64x32 (4x2 WMMA).
// Double-buffered LDS; async global->LDS copies when the toolchain has them.
// ---------------------------------------------------------------------------
#define BM  128
#define BN  128
#define BK  32
#define LDT 40   // padded LDS row stride, f16 elements

__device__ __forceinline__ void stage_tile(const _Float16* __restrict__ A,
                                           const _Float16* __restrict__ B,
                                           _Float16* sA, _Float16* sB,
                                           int t, int m0, int n0, int K, int k0) {
#pragma unroll
  for (int j = 0; j < 2; ++j) {
    int c  = t + j * 256;       // 512 chunks of 16B per tile
    int r  = c >> 2;
    int ch = c & 3;
    const _Float16* ga = A + (size_t)(m0 + r) * K + k0 + ch * 8;
    const _Float16* gb = B + (size_t)(n0 + r) * K + k0 + ch * 8;
    _Float16* la = sA + r * LDT + ch * 8;
    _Float16* lb = sB + r * LDT + ch * 8;
#if HAS_ASYNC
    __builtin_amdgcn_global_load_async_to_lds_b128((v4i*)(void*)ga, (v4i*)(void*)la, 0, 0);
    __builtin_amdgcn_global_load_async_to_lds_b128((v4i*)(void*)gb, (v4i*)(void*)lb, 0, 0);
#else
    *(v8h*)la = *(const v8h*)ga;
    *(v8h*)lb = *(const v8h*)gb;
#endif
  }
}

__global__ void __launch_bounds__(256) gemm_kernel(const _Float16* __restrict__ A,
                                                   const _Float16* __restrict__ B,
                                                   const float* __restrict__ bias,
                                                   float* __restrict__ out,
                                                   int M, int N, int K) {
  __shared__ _Float16 sA[2][BM * LDT];
  __shared__ _Float16 sB[2][BN * LDT];

  const int t    = threadIdx.x;
  const int lane = t & 31;
  const int wave = t >> 5;        // 0..7
  const int wm   = wave >> 2;     // 0..1 : which 64-row half
  const int wn   = wave & 3;      // 0..3 : which 32-col quarter
  const int lr   = lane & 15;
  const int hi   = lane >> 4;     // 0/1 : K/M half select

  const int m0 = blockIdx.y * BM;
  const int n0 = blockIdx.x * BN;

  v8f acc[4][2];
#pragma unroll
  for (int mi = 0; mi < 4; ++mi)
#pragma unroll
    for (int ni = 0; ni < 2; ++ni) {
      v8f z = {};
      acc[mi][ni] = z;
    }

  union U { v16h v; v8h h[2]; };

  // prologue: stage k0 = 0 into buffer 0
  stage_tile(A, B, sA[0], sB[0], t, m0, n0, K, 0);

  int buf = 0;
  for (int k0 = 0; k0 < K; k0 += BK) {
    const bool more = (k0 + BK) < K;
    // stage next slab into the other buffer (overlaps with compute below)
    if (more) stage_tile(A, B, sA[buf ^ 1], sB[buf ^ 1], t, m0, n0, K, k0 + BK);
#if HAS_ASYNC
    // 4 newly-issued chunks may be outstanding; older 4 (current buffer) done.
    if (more) __builtin_amdgcn_s_wait_asynccnt(4);
    else      __builtin_amdgcn_s_wait_asynccnt(0);
#endif
    __syncthreads();

    const _Float16* cA = sA[buf];
    const _Float16* cB = sB[buf];

    v16h afrag[4], bfrag[2];
#pragma unroll
    for (int mi = 0; mi < 4; ++mi) {
      int r = wm * 64 + mi * 16 + lr;
      U u;
      u.h[0] = *(const v8h*)(cA + r * LDT + hi * 8);        // K = hk..hk+7
      u.h[1] = *(const v8h*)(cA + r * LDT + hi * 8 + 16);   // K = hk+16..hk+23
      afrag[mi] = u.v;
    }
#pragma unroll
    for (int ni = 0; ni < 2; ++ni) {
      int n = wn * 32 + ni * 16 + lr;
      U u;
      u.h[0] = *(const v8h*)(cB + n * LDT + hi * 8);        // K = hk..hk+7
      u.h[1] = *(const v8h*)(cB + n * LDT + hi * 8 + 16);   // K = hk+16..hk+23
      bfrag[ni] = u.v;
    }

#pragma unroll
    for (int mi = 0; mi < 4; ++mi)
#pragma unroll
      for (int ni = 0; ni < 2; ++ni)
        acc[mi][ni] = __builtin_amdgcn_wmma_f32_16x16x32_f16(
            false, afrag[mi], false, bfrag[ni],
            (short)0, acc[mi][ni], false, false);

    // all waves done reading this buffer before it is overwritten next iter+1
    __syncthreads();
    buf ^= 1;
  }

  // epilogue: bias add + store (C/D layout: lane=N, vgpr=M, hi half -> M+8)
#pragma unroll
  for (int ni = 0; ni < 2; ++ni) {
    int n = n0 + wn * 32 + ni * 16 + lr;
    float bv = bias[n];
#pragma unroll
    for (int mi = 0; mi < 4; ++mi) {
      int mbase = m0 + wm * 64 + mi * 16 + hi * 8;
#pragma unroll
      for (int v = 0; v < 8; ++v) {
        out[(size_t)(mbase + v) * N + n] = acc[mi][ni][v] + bv;
      }
    }
  }
}

// ---------------------------------------------------------------------------
extern "C" void kernel_launch(void* const* d_in, const int* in_sizes, int n_in,
                              void* d_out, int out_size, void* d_ws, size_t ws_size,
                              hipStream_t stream) {
  const float* x      = (const float*)d_in[0];
  const int*   codes  = (const int*)d_in[1];
  const float* grid   = (const float*)d_in[2];
  const float* scales = (const float*)d_in[3];
  const float* bias   = (const float*)d_in[4];
  float* out = (float*)d_out;

  const int in_f   = IN_F;
  const int tokens = in_sizes[0] / in_f;   // 8192
  const int out_f  = in_sizes[4];          // 4096

  // workspace layout: xh (tokens*in_f f16) | Wd (out_f*in_f f16)  ~= 96 MB
  _Float16* xh = (_Float16*)d_ws;
  _Float16* Wd = xh + (size_t)tokens * in_f;

  had_kernel<<<tokens * (in_f / HAD), 256, 0, stream>>>(x, xh);
  dequant_kernel<<<(out_f * (in_f / 2)) / 256, 256, 0, stream>>>(codes, grid,
                                                                 scales, Wd);
  dim3 g(out_f / BN, tokens / BM);
  gemm_kernel<<<g, 256, 0, stream>>>(xh, Wd, bias, out, tokens, out_f, in_f);
}
